// Attention_39676907888265
// MI455X (gfx1250) — compile-verified, matrix operand order checked by reading, not run on hardware
//
#include <hip/hip_runtime.h>

// ---------------- problem constants ----------------
#define D_MODEL      2048
#define NUM_HEADS    16
#define QUERY_GROUPS 4
#define HEAD_DIM     128
#define BATCH        4
#define SEQ          2048
#define QKV_OUT      3072          // 16*128 + 2*4*128
#define KV_SZ        512
#define SCALE_F      0.08838834764831845f
#define NEG_INF_F    (-1000000000.0f)
#define LOG_THETA    9.2103403719761836f   // ln(10000)

typedef __attribute__((ext_vector_type(16))) __bf16 v16bf;
typedef __attribute__((ext_vector_type(8)))  float  v8f;

// -------- bf16 helpers (raw ushort storage, RNE convert) --------
__device__ __forceinline__ unsigned short f32_to_bf16(float f) {
  unsigned int u = __float_as_uint(f);
  u += 0x7fffu + ((u >> 16) & 1u);
  return (unsigned short)(u >> 16);
}
__device__ __forceinline__ float bf16_to_f32(unsigned short s) {
  return __uint_as_float(((unsigned int)s) << 16);
}

// ---------------------------------------------------------------
// gfx1250 async global->LDS copy (ASYNCcnt tracked), 16B per lane.
// ---------------------------------------------------------------
__device__ __forceinline__ void async_copy_b128(unsigned lds_off, const void* g) {
  asm volatile("global_load_async_to_lds_b128 %0, %1, off"
               :
               : "v"(lds_off), "v"((unsigned long long)(size_t)g)
               : "memory");
}
__device__ __forceinline__ void wait_async_le8() {
  asm volatile("s_wait_asynccnt 0x8" ::: "memory");
}
__device__ __forceinline__ void wait_async_le2() {
  asm volatile("s_wait_asynccnt 0x2" ::: "memory");
}
__device__ __forceinline__ void wait_async_0() {
  asm volatile("s_wait_asynccnt 0x0" ::: "memory");
}

// ---------------------------------------------------------------
// Fragment loader for V_WMMA_F32_16X16X32_BF16 (ISA 7.12.2 layout):
// lane L: row/col = L%16, two contiguous 8-elem K chunks at
// kb=(L/16)*8 and kb+16.  Works for global or LDS pointers.
// ---------------------------------------------------------------
__device__ __forceinline__ v16bf load_frag_bf16(const unsigned short* base, int ld) {
  const int lane = threadIdx.x & 31;
  const int row  = lane & 15;
  const int kb   = (lane >> 4) << 3;
  const unsigned short* p = base + (size_t)row * ld + kb;
  union { v16bf v; uint4 q[2]; } u;
  u.q[0] = *(const uint4*)(p);
  u.q[1] = *(const uint4*)(p + 16);
  return u.v;
}

__device__ __forceinline__ v8f wmma_bf16(v16bf a, v16bf b, v8f c) {
  return __builtin_amdgcn_wmma_f32_16x16x32_bf16(
      false, a, false, b, (short)0, c, false, false);
}

// ---------------------------------------------------------------
// elementwise f32 -> bf16 pack (vector of 4)
// ---------------------------------------------------------------
__global__ void cvt_bf16_kernel(const float* __restrict__ src,
                                unsigned short* __restrict__ dst, int n4) {
  const int i = blockIdx.x * blockDim.x + threadIdx.x;
  if (i >= n4) return;
  const float4 f = ((const float4*)src)[i];
  uint2 o;
  o.x = (unsigned)f32_to_bf16(f.x) | ((unsigned)f32_to_bf16(f.y) << 16);
  o.y = (unsigned)f32_to_bf16(f.z) | ((unsigned)f32_to_bf16(f.w) << 16);
  ((uint2*)dst)[i] = o;
}

// ---------------------------------------------------------------
// GEMM: C[M,N] = A[M,K] * W[N,K]^T, both bf16 K-contiguous row-major.
// Block = 128 threads (4 waves): the 64x32 W tile is staged in LDS
// via async copy (double-buffered, shared by all 4 waves); each wave
// computes its own 16-row M strip against the shared tile.
// ---------------------------------------------------------------
template <bool OUT_BF16>
__global__ __launch_bounds__(128, 1)
void gemm_wmma(const unsigned short* __restrict__ A,
               const unsigned short* __restrict__ W,
               void* __restrict__ Cv, int M, int N, int K) {
  __shared__ unsigned short Wtile[2][64][32];   // 8 KB, double-buffered

  const int tid   = threadIdx.x;
  const int wave  = tid >> 5;
  const int lane  = tid & 31;
  const int tileN = blockIdx.x * 64;
  const int tileM = blockIdx.y * 64 + wave * 16;

  const unsigned short* Abase = A + (size_t)tileM * K;
  const unsigned short* Wbase = W + (size_t)tileN * K;

  // async geometry: 4KB tile = 256 x 16B chunks, 2 per thread
  unsigned cOff[2], wEl[2];
#pragma unroll
  for (int j = 0; j < 2; ++j) {
    const unsigned c = tid + 128u * j;
    cOff[j] = c * 16u;                                 // LDS byte offset
    wEl[j]  = (c >> 2) * (unsigned)K + (c & 3u) * 8u;  // W row-major [N][K]
  }
  const unsigned wtB[2] = {(unsigned)(size_t)&Wtile[0][0][0],
                           (unsigned)(size_t)&Wtile[1][0][0]};

  auto issue_w = [&](int k, int buf) {
#pragma unroll
    for (int j = 0; j < 2; ++j)
      async_copy_b128(wtB[buf] + cOff[j], Wbase + (size_t)k + wEl[j]);
  };

  v8f acc[4] = {v8f{}, v8f{}, v8f{}, v8f{}};

  issue_w(0, 0);
  for (int k = 0, it = 0; k < K; k += 32, ++it) {
    const int buf = it & 1;
    __syncthreads();                      // prior readers of buf^1 done
    if (k + 32 < K) {
      issue_w(k + 32, buf ^ 1);           // 2 async ops per wave
      wait_async_le2();                   // tile `it` landed (in-order)
    } else {
      wait_async_0();
    }
    __syncthreads();                      // all waves' copies visible

    if (k + 64 < K) __builtin_prefetch(Abase + k + 64, 0, 3);
    const v16bf a = load_frag_bf16(Abase + k, K);
    const unsigned short* Wt = &Wtile[buf][0][0];
#pragma unroll
    for (int j = 0; j < 4; ++j) {
      const v16bf b = load_frag_bf16(Wt + (size_t)(j * 16) * 32, 32);
      acc[j] = wmma_bf16(a, b, acc[j]);
    }
  }

  const int col = lane & 15;
  const int rOf = (lane >> 4) << 3;
#pragma unroll
  for (int j = 0; j < 4; ++j) {
#pragma unroll
    for (int r = 0; r < 8; ++r) {
      const size_t idx = (size_t)(tileM + rOf + r) * N + (tileN + j * 16 + col);
      const float v = acc[j][r];
      if (OUT_BF16) ((unsigned short*)Cv)[idx] = f32_to_bf16(v);
      else          ((float*)Cv)[idx]          = v;
    }
  }
}

// ---------------------------------------------------------------
// RoPE + split qkv[b,t,3072] (bf16) into Qr[b,h,t,d], Kr[b,g,t,d],
// Vt[b,g,d,t] (V transposed so PV B-operand is K-contiguous).
// ---------------------------------------------------------------
__global__ void rope_split_kernel(const unsigned short* __restrict__ qkv,
                                  unsigned short* __restrict__ Qr,
                                  unsigned short* __restrict__ Kr,
                                  unsigned short* __restrict__ Vt) {
  const int row = blockIdx.x;
  const int b = row >> 11;
  const int t = row & (SEQ - 1);
  const unsigned short* src = qkv + (size_t)row * QKV_OUT;
  const float ft = (float)t;

  for (int idx = threadIdx.x; idx < NUM_HEADS * 64; idx += blockDim.x) {
    const int h = idx >> 6, i = idx & 63;
    const float x1 = bf16_to_f32(src[h * HEAD_DIM + i]);
    const float x2 = bf16_to_f32(src[h * HEAD_DIM + 64 + i]);
    const float inv = __expf(-((float)(2 * i) * (1.0f / 128.0f)) * LOG_THETA);
    float s, c; __sincosf(ft * inv, &s, &c);
    const size_t base = (((size_t)(b * NUM_HEADS + h)) * SEQ + t) * HEAD_DIM;
    Qr[base + i]      = f32_to_bf16(x1 * c - x2 * s);
    Qr[base + 64 + i] = f32_to_bf16(x2 * c + x1 * s);
  }
  for (int idx = threadIdx.x; idx < QUERY_GROUPS * 64; idx += blockDim.x) {
    const int g = idx >> 6, i = idx & 63;
    const float x1 = bf16_to_f32(src[D_MODEL + g * HEAD_DIM + i]);
    const float x2 = bf16_to_f32(src[D_MODEL + g * HEAD_DIM + 64 + i]);
    const float inv = __expf(-((float)(2 * i) * (1.0f / 128.0f)) * LOG_THETA);
    float s, c; __sincosf(ft * inv, &s, &c);
    const size_t base = (((size_t)(b * QUERY_GROUPS + g)) * SEQ + t) * HEAD_DIM;
    Kr[base + i]      = f32_to_bf16(x1 * c - x2 * s);
    Kr[base + 64 + i] = f32_to_bf16(x2 * c + x1 * s);
  }
  for (int idx = threadIdx.x; idx < QUERY_GROUPS * HEAD_DIM; idx += blockDim.x) {
    const int g = idx >> 7, d = idx & 127;
    Vt[(((size_t)(b * QUERY_GROUPS + g)) * HEAD_DIM + d) * SEQ + t] =
        src[D_MODEL + KV_SZ + g * HEAD_DIM + d];
  }
}

// ---------------------------------------------------------------
// Flash attention. Block = 128 threads (4 waves) = 64 query rows for
// one (b,h). K/V 32-key tiles double-buffered in LDS via
// global_load_async_to_lds_b128, shared by all 4 waves.
// ---------------------------------------------------------------
__global__ __launch_bounds__(128, 1)
void attn_kernel(const unsigned short* __restrict__ Qr,
                 const unsigned short* __restrict__ Kr,
                 const unsigned short* __restrict__ Vt,
                 const unsigned char* __restrict__ mask,
                 unsigned short* __restrict__ attn_out) {
  __shared__ unsigned short Ktile[2][32][HEAD_DIM];   // 16 KB
  __shared__ unsigned short Vtile[2][HEAD_DIM][32];   // 16 KB
  __shared__ float         Plds[4][16][32];           //  8 KB
  __shared__ unsigned char Mlds[SEQ];                 //  2 KB

  const int tid  = threadIdx.x;
  const int wave = tid >> 5;
  const int lane = tid & 31;
  const int col  = lane & 15;
  const int h = blockIdx.y, b = blockIdx.z, g = h >> 2;
  const int qbase = blockIdx.x * 64 + wave * 16;

  const unsigned short* Qbase = Qr + (((size_t)(b * NUM_HEADS + h)) * SEQ + qbase) * HEAD_DIM;
  const unsigned short* Kbase = Kr + ((size_t)(b * QUERY_GROUPS + g)) * SEQ * HEAD_DIM;
  const unsigned short* Vbase = Vt + ((size_t)(b * QUERY_GROUPS + g)) * HEAD_DIM * SEQ;
  const unsigned char*  mrow_base = mask + (size_t)b * SEQ;

  for (int i = tid; i < SEQ; i += 128) Mlds[i] = mrow_base[i];

  // per-thread async-copy geometry: 512 x 16B chunks per 8KB tile,
  // 4 K-chunks + 4 V-chunks per thread -> 8 async ops per WAVE per pair
  unsigned cOff[4], kEl[4], vEl[4];
#pragma unroll
  for (int j = 0; j < 4; ++j) {
    const unsigned c = tid + 128u * j;
    cOff[j] = c * 16u;
    kEl[j]  = (c >> 4) * HEAD_DIM + (c & 15u) * 8u;    // K: [key][d]
    vEl[j]  = (c >> 2) * SEQ + (c & 3u) * 8u;          // V: [d][t]
  }
  const unsigned ktB[2] = {(unsigned)(size_t)&Ktile[0][0][0],
                           (unsigned)(size_t)&Ktile[1][0][0]};
  const unsigned vtB[2] = {(unsigned)(size_t)&Vtile[0][0][0],
                           (unsigned)(size_t)&Vtile[1][0][0]};

  auto issue_tiles = [&](int kb, int buf) {
#pragma unroll
    for (int j = 0; j < 4; ++j)
      async_copy_b128(ktB[buf] + cOff[j], Kbase + (size_t)kb * HEAD_DIM + kEl[j]);
#pragma unroll
    for (int j = 0; j < 4; ++j)
      async_copy_b128(vtB[buf] + cOff[j], Vbase + kb + vEl[j]);
  };

  v16bf qf[4];
#pragma unroll
  for (int c = 0; c < 4; ++c) qf[c] = load_frag_bf16(Qbase + c * 32, HEAD_DIM);

  v8f Oacc[8];
#pragma unroll
  for (int n = 0; n < 8; ++n) Oacc[n] = v8f{};
  float mrow[8], lrow[8];
#pragma unroll
  for (int r = 0; r < 8; ++r) { mrow[r] = -3.0e38f; lrow[r] = 0.0f; }

  issue_tiles(0, 0);

  for (int kb = 0, it = 0; kb < SEQ; kb += 32, ++it) {
    const int buf = it & 1;
    __syncthreads();
    if (kb + 32 < SEQ) {
      issue_tiles(kb + 32, buf ^ 1);
      wait_async_le8();
    } else {
      wait_async_0();
    }
    __syncthreads();

    const unsigned short* Kt = &Ktile[buf][0][0];
    const unsigned short* Vl = &Vtile[buf][0][0];

    v8f S0 = v8f{}, S1 = v8f{};
#pragma unroll
    for (int c = 0; c < 4; ++c) {
      S0 = wmma_bf16(qf[c], load_frag_bf16(Kt + c * 32, HEAD_DIM), S0);
      S1 = wmma_bf16(qf[c], load_frag_bf16(Kt + 16 * HEAD_DIM + c * 32, HEAD_DIM), S1);
    }

    const bool mk0 = Mlds[kb + col]      != 0;
    const bool mk1 = Mlds[kb + 16 + col] != 0;
#pragma unroll
    for (int r = 0; r < 8; ++r) {
      S0[r] = mk0 ? S0[r] * SCALE_F : NEG_INF_F;
      S1[r] = mk1 ? S1[r] * SCALE_F : NEG_INF_F;
    }

#pragma unroll
    for (int r = 0; r < 8; ++r) {
      float cm = fmaxf(S0[r], S1[r]);
#pragma unroll
      for (int off = 8; off >= 1; off >>= 1) cm = fmaxf(cm, __shfl_xor(cm, off, 32));
      const float nm    = fmaxf(mrow[r], cm);
      const float alpha = __expf(mrow[r] - nm);
      mrow[r] = nm;
      const float p0 = __expf(S0[r] - nm);
      const float p1 = __expf(S1[r] - nm);
      float ps = p0 + p1;
#pragma unroll
      for (int off = 8; off >= 1; off >>= 1) ps += __shfl_xor(ps, off, 32);
      lrow[r] = lrow[r] * alpha + ps;
#pragma unroll
      for (int n = 0; n < 8; ++n) Oacc[n][r] *= alpha;
      const int m = r + ((lane >> 4) << 3);
      Plds[wave][m][col]      = p0;
      Plds[wave][m][16 + col] = p1;
    }
    __builtin_amdgcn_s_barrier();        // wave-local LDS ordering

    union { v16bf v; unsigned short s[16]; } pu;
    {
      const int prow = lane & 15;
      const int pkb  = (lane >> 4) << 3;
#pragma unroll
      for (int e = 0; e < 8; ++e) {
        pu.s[e]     = f32_to_bf16(Plds[wave][prow][pkb + e]);
        pu.s[e + 8] = f32_to_bf16(Plds[wave][prow][pkb + 16 + e]);
      }
    }

#pragma unroll
    for (int n = 0; n < 8; ++n) {
      const v16bf vf = load_frag_bf16(Vl + (size_t)(n * 16) * 32, 32);
      Oacc[n] = wmma_bf16(pu.v, vf, Oacc[n]);
    }
  }

#pragma unroll
  for (int r = 0; r < 8; ++r) {
    const int m = r + ((lane >> 4) << 3);
    const int q = qbase + m;
    const float rl = lrow[r] > 0.0f ? 1.0f / lrow[r] : 0.0f;
    const float s  = rl * (Mlds[q] ? 1.0f : 0.0f);
#pragma unroll
    for (int n = 0; n < 8; ++n) {
      attn_out[((size_t)(b * SEQ + q)) * D_MODEL + h * HEAD_DIM + n * 16 + col] =
          f32_to_bf16(Oacc[n][r] * s);
    }
  }
}

// ---------------------------------------------------------------
extern "C" void kernel_launch(void* const* d_in, const int* in_sizes, int n_in,
                              void* d_out, int out_size, void* d_ws, size_t ws_size,
                              hipStream_t stream) {
  const float*         x     = (const float*)d_in[0];
  const unsigned char* pmask = (const unsigned char*)d_in[1];
  const float*         qkv_w = (const float*)d_in[2];
  const float*         o_w   = (const float*)d_in[3];
  float*               out   = (float*)d_out;

  const int M = BATCH * SEQ;                       // 8192
  const size_t nX  = (size_t)M * D_MODEL;
  const size_t nWq = (size_t)QKV_OUT * D_MODEL;
  const size_t nWo = (size_t)D_MODEL * D_MODEL;

  unsigned short* x_bf   = (unsigned short*)d_ws;
  unsigned short* wq_bf  = x_bf  + nX;
  unsigned short* wo_bf  = wq_bf + nWq;
  unsigned short* qkv_ws = wo_bf + nWo;
  unsigned short* Qr     = qkv_ws + (size_t)M * QKV_OUT;
  unsigned short* Kr     = Qr + (size_t)BATCH * NUM_HEADS * SEQ * HEAD_DIM;
  unsigned short* Vt     = Kr + (size_t)BATCH * QUERY_GROUPS * SEQ * HEAD_DIM;
  unsigned short* attn   = Vt + (size_t)BATCH * QUERY_GROUPS * SEQ * HEAD_DIM;

  cvt_bf16_kernel<<<(int)(nX  / 4 + 255) / 256, 256, 0, stream>>>(x,     x_bf,  (int)(nX  / 4));
  cvt_bf16_kernel<<<(int)(nWq / 4 + 255) / 256, 256, 0, stream>>>(qkv_w, wq_bf, (int)(nWq / 4));
  cvt_bf16_kernel<<<(int)(nWo / 4 + 255) / 256, 256, 0, stream>>>(o_w,   wo_bf, (int)(nWo / 4));

  // 1) qkv = x @ qkv_w^T   (bf16 in, bf16 out)
  gemm_wmma<true><<<dim3(QKV_OUT / 64, M / 64), 128, 0, stream>>>(
      x_bf, wq_bf, (void*)qkv_ws, M, QKV_OUT, D_MODEL);

  // 2) RoPE + split + V-transpose
  rope_split_kernel<<<M, 256, 0, stream>>>(qkv_ws, Qr, Kr, Vt);

  // 3) flash attention (async LDS double-buffered K/V)
  attn_kernel<<<dim3(SEQ / 64, NUM_HEADS, BATCH), 128, 0, stream>>>(
      Qr, Kr, Vt, pmask, attn);

  // 4) out = attn @ o_w^T  (bf16 in, f32 out)
  gemm_wmma<false><<<dim3(D_MODEL / 64, M / 64), 128, 0, stream>>>(
      attn, wo_bf, (void*)out, M, D_MODEL, D_MODEL);
}
